// GraphConvolution_39573828666028
// MI455X (gfx1250) — matrix operand order repaired and checked by reading, not compiled, not run on hardware
//
#include <hip/hip_runtime.h>

// ---------------------------------------------------------------------------
// GraphConvolution forward for MI455X (gfx1250, wave32, bf16 WMMA + async-LDS)
//   deg = rowwise count(adj > 0)                      (fused into GEMM1)
//   agg = (1/deg) * (adj @ v_feature)                 (GEMM1, bf16 WMMA)
//   out = relu(concat(agg, u_feature) @ weight)       (GEMM2, bf16 WMMA)
// v_feature / weight are pre-converted to bf16 and pre-swizzled into the LDS
// B-fragment image, so per-K-step staging is a contiguous 16KB async copy.
// ---------------------------------------------------------------------------

#define N_U   8192
#define N_V   8192
#define IN_U  256
#define IN_V  256
#define OUT_U 256
#define KTOT2 (IN_V + IN_U)       // 512

#define KSTEP 32                  // WMMA K depth (bf16)
#define WAVES 4
#define MTILE (WAVES * 16)        // 64 rows per block
#define PANEL_ELEMS (KSTEP * 256) // 8192 bf16 = 16 KB per K-panel

typedef __attribute__((ext_vector_type(16))) __bf16 v16bf;
typedef __attribute__((ext_vector_type(8)))  __bf16 v8bf;
typedef __attribute__((ext_vector_type(8)))  float  v8f;
typedef __attribute__((ext_vector_type(4)))  float  f32x4;
typedef __attribute__((ext_vector_type(4)))  unsigned int u32x4;
typedef __attribute__((ext_vector_type(4)))  int i32x4;

// pack two f32 -> two bf16 in one dword (lo, hi)
__device__ __forceinline__ unsigned int pk_bf16x2(float lo, float hi) {
  __bf16 a = (__bf16)lo, b = (__bf16)hi;
  unsigned short ua = __builtin_bit_cast(unsigned short, a);
  unsigned short ub = __builtin_bit_cast(unsigned short, b);
  return (unsigned int)ua | ((unsigned int)ub << 16);
}

// adj in {0,1,2}: nonzero count of a float4 via integer umin (no canonicalize)
__device__ __forceinline__ unsigned int nz4(f32x4 v) {
  u32x4 b = __builtin_bit_cast(u32x4, v);
  return min(b.x, 1u) + min(b.y, 1u) + min(b.z, 1u) + min(b.w, 1u);
}

// ---- CDNA5 async global->LDS copy (ASYNCcnt) with portable fallback -------
__device__ __forceinline__ void async_copy16(const void* g, void* l) {
#if __has_builtin(__builtin_amdgcn_global_load_async_to_lds_b128)
  __builtin_amdgcn_global_load_async_to_lds_b128(
      (__attribute__((address_space(1))) i32x4*)(uintptr_t)g,
      (__attribute__((address_space(3))) i32x4*)(unsigned int)(uintptr_t)l,
      0, 0);
#else
  *(u32x4*)l = *(const u32x4*)g;   // b128 load + ds_store_b128 fallback
#endif
}

__device__ __forceinline__ void wait_async() {
#if __has_builtin(__builtin_amdgcn_global_load_async_to_lds_b128)
#  if __has_builtin(__builtin_amdgcn_s_wait_asynccnt)
  __builtin_amdgcn_s_wait_asynccnt(0);
#  else
  asm volatile("s_wait_asynccnt 0x0" ::: "memory");
#  endif
#endif
}

// Copy one 16KB fragment-ordered panel from global to LDS (128 threads).
__device__ __forceinline__ void stage_panel(const __bf16* gpanel, __bf16* lpanel,
                                            int tid) {
  #pragma unroll
  for (int j = 0; j < 8; ++j) {
    int off = (j * 128 + tid) * 8;           // in __bf16 elements (16B chunks)
    async_copy16(gpanel + off, lpanel + off);
  }
}

// ---------------------------------------------------------------------------
// Prologue: convert f32 [nrows][256] -> bf16 panels in B-fragment order:
//   element (k, c) -> dst[(k/32)*8192 + ((c>>4)*32 + (c&15) + 16*(((k&31)>>3)&1))*16
//                        + (k&7) + 8*((k&31)>>4)]
// One thread per (row-pair, col-quad); paired rows pack into dword stores.
// ---------------------------------------------------------------------------
__global__ __launch_bounds__(256) void swizzle_bf16_panels(
    const float* __restrict__ src, __bf16* __restrict__ dst, int nrows)
{
  int id = blockIdx.x * 256 + threadIdx.x;
  int total = (nrows >> 1) * 64;
  if (id >= total) return;
  int k = (id >> 6) << 1;                    // even row
  int c = (id & 63) << 2;                    // column quad
  f32x4 v0 = *(const f32x4*)(src + (size_t)k * 256 + c);
  f32x4 v1 = *(const f32x4*)(src + (size_t)(k + 1) * 256 + c);
  int ks  = k >> 5;
  int kk  = k & 31;
  int nt  = c >> 4;
  int lb  = (c & 15) + (((kk >> 3) & 1) << 4);
  int idx = (kk & 7) + ((kk >> 4) << 3);     // even
  __bf16* base = dst + (size_t)ks * PANEL_ELEMS + (size_t)(nt * 32) * 16;
  *(unsigned int*)&base[(lb + 0) * 16 + idx] = pk_bf16x2(v0.x, v1.x);
  *(unsigned int*)&base[(lb + 1) * 16 + idx] = pk_bf16x2(v0.y, v1.y);
  *(unsigned int*)&base[(lb + 2) * 16 + idx] = pk_bf16x2(v0.z, v1.z);
  *(unsigned int*)&base[(lb + 3) * 16 + idx] = pk_bf16x2(v0.w, v1.w);
}

// ---------------------------------------------------------------------------
// Kernel 1: agg_bf16 = inv_deg * (adj @ v_feature); deg fused; adj read once.
// ---------------------------------------------------------------------------
__global__ __launch_bounds__(128) void gemm1_deg_scale(
    const float* __restrict__ adj,
    const __bf16* __restrict__ vswz,         // 256 fragment-ordered panels
    __bf16* __restrict__ aggb)
{
  __shared__ __bf16 ldsB[2][16][32][16];     // 2 x 16 KB

  const int tid   = threadIdx.x;
  const int wave  = tid >> 5;
  const int lane  = tid & 31;
  const int lrow  = lane & 15;
  const int khalf = lane >> 4;
  const int mbase = blockIdx.x * MTILE + wave * 16;
  const size_t arow = (size_t)(mbase + lrow) * (size_t)N_V;

  v8f acc[16];
  #pragma unroll
  for (int n = 0; n < 16; ++n) {
    v8f z = {0.f,0.f,0.f,0.f,0.f,0.f,0.f,0.f};
    acc[n] = z;
  }
  unsigned int cnt = 0;   // nonzero count over this lane's share of its row

  stage_panel(vswz, &ldsB[0][0][0][0], tid);
  wait_async();
  __syncthreads();

  const int NSTEP = N_V / KSTEP;             // 256
  for (int ks = 0; ks < NSTEP; ++ks) {
    const int buf = ks & 1;
    if (ks + 1 < NSTEP)
      stage_panel(vswz + (size_t)(ks + 1) * PANEL_ELEMS,
                  &ldsB[buf ^ 1][0][0][0], tid);

    // ---- A fragment: 16x32 bf16 (lane holds K {h*8..+7, 16+h*8..+7}) ----
    const float* p = adj + arow + (size_t)(ks * KSTEP + khalf * 8);
    f32x4 a0 = *(const f32x4*)(p + 0);
    f32x4 a1 = *(const f32x4*)(p + 4);
    f32x4 a2 = *(const f32x4*)(p + 16);
    f32x4 a3 = *(const f32x4*)(p + 20);
    cnt += nz4(a0) + nz4(a1) + nz4(a2) + nz4(a3);
    v16bf af;
    af[0]=(__bf16)a0.x; af[1]=(__bf16)a0.y; af[2]=(__bf16)a0.z; af[3]=(__bf16)a0.w;
    af[4]=(__bf16)a1.x; af[5]=(__bf16)a1.y; af[6]=(__bf16)a1.z; af[7]=(__bf16)a1.w;
    af[8]=(__bf16)a2.x; af[9]=(__bf16)a2.y; af[10]=(__bf16)a2.z; af[11]=(__bf16)a2.w;
    af[12]=(__bf16)a3.x; af[13]=(__bf16)a3.y; af[14]=(__bf16)a3.z; af[15]=(__bf16)a3.w;

    #pragma unroll
    for (int n = 0; n < 16; ++n) {
      v16bf bfrag = *(const v16bf*)&ldsB[buf][n][lane][0];
      acc[n] = __builtin_amdgcn_wmma_f32_16x16x32_bf16(
          false, af, false, bfrag, (short)0, acc[n], false, false);
    }
    wait_async();
    __syncthreads();
  }

  // ---- deg -> inv_deg, broadcast to C-layout rows ----
  cnt += __shfl_xor((int)cnt, 16, 32);       // combine the two K-halves
  float degf = (float)cnt;
  float invd = (cnt > 0u) ? (1.0f / degf) : 0.0f;     // row (lane&15)
  float s[8];
  #pragma unroll
  for (int g = 0; g < 8; ++g)
    s[g] = __shfl(invd, ((lane >> 4) << 3) + g, 32);  // row g + 8*(lane>>4)

  #pragma unroll
  for (int n = 0; n < 16; ++n) {
    #pragma unroll
    for (int g = 0; g < 8; ++g) {
      int r = mbase + g + ((lane >> 4) << 3);
      int c = n * 16 + (lane & 15);
      aggb[(size_t)r * IN_V + c] = (__bf16)(acc[n][g] * s[g]);
    }
  }
}

// ---------------------------------------------------------------------------
// Kernel 2: out = relu([agg | u] @ W), K = 512 (0..255 agg bf16, 256..511 u)
// ---------------------------------------------------------------------------
__global__ __launch_bounds__(128) void gemm2_relu(
    const __bf16* __restrict__ aggb,
    const float*  __restrict__ ufeat,
    const __bf16* __restrict__ wswz,         // 16 fragment-ordered panels
    float* __restrict__ out)
{
  __shared__ __bf16 ldsB[2][16][32][16];

  const int tid   = threadIdx.x;
  const int wave  = tid >> 5;
  const int lane  = tid & 31;
  const int lrow  = lane & 15;
  const int khalf = lane >> 4;
  const int mbase = blockIdx.x * MTILE + wave * 16;
  const int myrow = mbase + lrow;

  v8f acc[16];
  #pragma unroll
  for (int n = 0; n < 16; ++n) {
    v8f z = {0.f,0.f,0.f,0.f,0.f,0.f,0.f,0.f};
    acc[n] = z;
  }

  stage_panel(wswz, &ldsB[0][0][0][0], tid);
  wait_async();
  __syncthreads();

  const int NSTEP = KTOT2 / KSTEP;           // 16
  for (int ks = 0; ks < NSTEP; ++ks) {
    const int buf = ks & 1;
    if (ks + 1 < NSTEP)
      stage_panel(wswz + (size_t)(ks + 1) * PANEL_ELEMS,
                  &ldsB[buf ^ 1][0][0][0], tid);

    const int k0 = ks * KSTEP;
    v16bf af;
    if (k0 < IN_V) {
      const __bf16* pb = aggb + (size_t)myrow * IN_V + (k0 + khalf * 8);
      v8bf c0 = *(const v8bf*)(pb);
      v8bf c1 = *(const v8bf*)(pb + 16);
      #pragma unroll
      for (int i = 0; i < 8; ++i) { af[i] = c0[i]; af[8 + i] = c1[i]; }
    } else {
      const float* p = ufeat + (size_t)myrow * IN_U + (k0 - IN_V + khalf * 8);
      f32x4 a0 = *(const f32x4*)(p + 0);
      f32x4 a1 = *(const f32x4*)(p + 4);
      f32x4 a2 = *(const f32x4*)(p + 16);
      f32x4 a3 = *(const f32x4*)(p + 20);
      af[0]=(__bf16)a0.x; af[1]=(__bf16)a0.y; af[2]=(__bf16)a0.z; af[3]=(__bf16)a0.w;
      af[4]=(__bf16)a1.x; af[5]=(__bf16)a1.y; af[6]=(__bf16)a1.z; af[7]=(__bf16)a1.w;
      af[8]=(__bf16)a2.x; af[9]=(__bf16)a2.y; af[10]=(__bf16)a2.z; af[11]=(__bf16)a2.w;
      af[12]=(__bf16)a3.x; af[13]=(__bf16)a3.y; af[14]=(__bf16)a3.z; af[15]=(__bf16)a3.w;
    }

    #pragma unroll
    for (int n = 0; n < 16; ++n) {
      v16bf bfrag = *(const v16bf*)&ldsB[buf][n][lane][0];
      acc[n] = __builtin_amdgcn_wmma_f32_16x16x32_bf16(
          false, af, false, bfrag, (short)0, acc[n], false, false);
    }
    wait_async();
    __syncthreads();
  }

  #pragma unroll
  for (int n = 0; n < 16; ++n) {
    #pragma unroll
    for (int g = 0; g < 8; ++g) {
      int r = mbase + g + ((lane >> 4) << 3);
      int c = n * 16 + (lane & 15);
      out[(size_t)r * OUT_U + c] = fmaxf(acc[n][g], 0.0f);
    }
  }
}

// ---------------------------------------------------------------------------
extern "C" void kernel_launch(void* const* d_in, const int* in_sizes, int n_in,
                              void* d_out, int out_size, void* d_ws, size_t ws_size,
                              hipStream_t stream) {
  const float* adj   = (const float*)d_in[0];
  const float* ufeat = (const float*)d_in[1];
  const float* vfeat = (const float*)d_in[2];
  const float* wgt   = (const float*)d_in[3];
  float* out = (float*)d_out;

  char* ws = (char*)d_ws;
  __bf16* vswz = (__bf16*)ws;                                  // 4 MB
  __bf16* wswz = (__bf16*)(ws + (size_t)4 * 1024 * 1024);      // 256 KB
  __bf16* aggb = (__bf16*)(ws + (size_t)4 * 1024 * 1024 + 256 * 1024); // 4 MB

  (void)in_sizes; (void)n_in; (void)out_size; (void)ws_size;

  // v_feature: (8192/2)*64 = 262144 work items; weight: (512/2)*64 = 16384
  swizzle_bf16_panels<<<dim3(1024), dim3(256), 0, stream>>>(vfeat, vswz, N_V);
  swizzle_bf16_panels<<<dim3(64),   dim3(256), 0, stream>>>(wgt,   wswz, KTOT2);
  gemm1_deg_scale<<<dim3(N_U / MTILE), dim3(128), 0, stream>>>(adj, vswz, aggb);
  gemm2_relu     <<<dim3(N_U / MTILE), dim3(128), 0, stream>>>(aggb, ufeat, wswz, out);
}
